// GNNModel_90933047591343
// MI455X (gfx1250) — matrix operand order repaired and checked by reading, not compiled, not run on hardware
//
#include <hip/hip_runtime.h>
#include <hip/hip_bf16.h>

typedef __attribute__((ext_vector_type(16))) _Float16 v16h;
typedef __attribute__((ext_vector_type(8)))  _Float16 v8h;
typedef __attribute__((ext_vector_type(8)))  float    v8f;

#define N_NODES   20736      // 256 puzzles * 81 cells
#define C_DIM     128
#define NEIGH     20
#define ATT_BLKS  2592       // N_NODES / 8 nodes-per-block

// ---------------------------------------------------------------------------
// Pack Wl|Wr (fp32 [8][128][128] each) into the CDNA5 WMMA B-fragment layout:
// flat fragment index f = ((l*16 + ct)*4 + ks)*32 + lane, 16 halves per lane.
// Per lane: col n = ct*16 + (lane&15); K = ks*32 + 8*(lane>=16) + (hh&7) + 16*(hh>=8).
// ---------------------------------------------------------------------------
__global__ __launch_bounds__(256)
void pack_w_kernel(const float* __restrict__ Wl, const float* __restrict__ Wr,
                   _Float16* __restrict__ wpack)
{
    const int idx  = blockIdx.x * 256 + threadIdx.x;   // 16384 fragments total
    const int lane = idx & 31;
    const int ks   = (idx >> 5) & 3;
    const int ct   = (idx >> 7) & 15;
    const int l    = idx >> 11;
    const int koff = (lane >> 4) * 8;
    const int n    = ct * 16 + (lane & 15);
    _Float16* dst  = wpack + (size_t)idx * 16;
    #pragma unroll
    for (int hh = 0; hh < 16; ++hh) {
        const int k = ks * 32 + koff + (hh & 7) + ((hh >> 3) << 4);
        const float v = (ct < 8)
            ? Wl[(size_t)l * 16384 + (size_t)k * 128 + n]
            : Wr[(size_t)l * 16384 + (size_t)k * 128 + (n - 128)];
        dst[hh] = (_Float16)v;
    }
}

// ---------------------------------------------------------------------------
// Embedding lookup: h[n][c] = embed_w[x[n]][c]; also emit f16 copy for WMMA.
// ---------------------------------------------------------------------------
__global__ __launch_bounds__(256)
void embed_kernel(const int* __restrict__ x, const float* __restrict__ ew,
                  float* __restrict__ h, _Float16* __restrict__ hf)
{
    const int idx = blockIdx.x * 256 + threadIdx.x;    // N*128
    const int n = idx >> 7, c = idx & 127;
    const float v = ew[x[n] * 128 + c];
    h[idx]  = v;
    hf[idx] = (_Float16)v;
}

// ---------------------------------------------------------------------------
// GEMM: xlr[N][256] = h_f16[N][128] @ [Wl | Wr] (+bias at store).
// Block = 128 threads = 4 waves; block owns a 16-row tile, each wave a 64-col
// group. 16 x v_wmma_f32_16x16x32_f16 per wave, A fragments straight from
// row-major global f16 (two 16B loads), B fragments from the packed layout
// (one contiguous 32B load per lane).
// ---------------------------------------------------------------------------
__global__ __launch_bounds__(128)
void gemm_xlr_kernel(const _Float16* __restrict__ hf,
                     const _Float16* __restrict__ wpack,
                     const float* __restrict__ bl, const float* __restrict__ br,
                     float* __restrict__ xlr, int layer)
{
    const int lane = threadIdx.x & 31;
    const int wave = threadIdx.x >> 5;        // col-group 0..3 (64 cols each)
    const int row0 = blockIdx.x * 16;
    const int arow = row0 + (lane & 15);
    const int kb   = (lane >> 4) * 8;         // A-fragment K sub-offset (ISA layout)

    v8f acc[4] = {};
    const _Float16* wb = wpack + (size_t)layer * 32768;

    #pragma unroll
    for (int ks = 0; ks < 4; ++ks) {
        const _Float16* ap = hf + (size_t)arow * 128 + ks * 32 + kb;
        const v8h alo = *(const v8h*)(ap);
        const v8h ahi = *(const v8h*)(ap + 16);
        v16h a;
        #pragma unroll
        for (int i = 0; i < 8; ++i) { a[i] = alo[i]; a[8 + i] = ahi[i]; }

        #pragma unroll
        for (int t = 0; t < 4; ++t) {
            const int ct = wave * 4 + t;
            const v16h b = *(const v16h*)(wb + (size_t)((ct * 4 + ks) * 32 + lane) * 16);
            acc[t] = __builtin_amdgcn_wmma_f32_16x16x32_f16(
                false, a, false, b, (short)0, acc[t], false, false);
        }
    }

    #pragma unroll
    for (int t = 0; t < 4; ++t) {
        const int ct  = wave * 4 + t;
        const int col = ct * 16 + (lane & 15);
        const float bias = (col < 128) ? bl[layer * 128 + col]
                                       : br[layer * 128 + col - 128];
        #pragma unroll
        for (int i = 0; i < 8; ++i) {
            const int row = row0 + i + ((lane >> 4) * 8);   // C/D ISA layout
            xlr[(size_t)row * 256 + col] = acc[t][i] + bias;
        }
    }
}

// ---------------------------------------------------------------------------
// Node-centric GATv2 attention + aggregation. One wave per destination node
// (exactly 20 neighbors; neighbor list of v == dst-list of src-group v by
// symmetry). Lane owns 4 channels (head = lane/8). Per-wave softmax in
// registers; fused block partial sums for the global LayerNorm statistics.
// ---------------------------------------------------------------------------
__global__ __launch_bounds__(256)
void attn_agg_kernel(const float* __restrict__ xlr, const int* __restrict__ dstArr,
                     const float* __restrict__ att, const float* __restrict__ conv_b,
                     float* __restrict__ agg, float* __restrict__ partials, int layer)
{
    __shared__ float s_sum[8], s_sq[8];
    const int lane = threadIdx.x & 31;
    const int wave = threadIdx.x >> 5;
    const int v    = blockIdx.x * 8 + wave;
    const int b    = v / 81;
    const int vloc = v - b * 81;
    const int ebase = b * 1620 + vloc * NEIGH;

    float xr_v[4], aw[4];
    #pragma unroll
    for (int j = 0; j < 4; ++j) {
        xr_v[j] = xlr[(size_t)v * 256 + 128 + lane * 4 + j];   // lin_r(x)[v]
        aw[j]   = att[layer * 128 + lane * 4 + j];
    }

    float logits[NEIGH];
    #pragma unroll 1
    for (int k = 0; k < NEIGH; ++k) {
        const int u = dstArr[ebase + k];
        float p = 0.f;
        #pragma unroll
        for (int j = 0; j < 4; ++j) {
            float e = xlr[(size_t)u * 256 + lane * 4 + j] + xr_v[j];
            e = (e > 0.f) ? e : 0.2f * e;                      // leaky_relu
            p += e * aw[j];
        }
        p += __shfl_xor(p, 1, 8);                              // reduce within head
        p += __shfl_xor(p, 2, 8);
        p += __shfl_xor(p, 4, 8);
        logits[k] = p;
    }

    float m = logits[0];
    #pragma unroll
    for (int k = 1; k < NEIGH; ++k) m = fmaxf(m, logits[k]);
    float ssum = 0.f;
    #pragma unroll
    for (int k = 0; k < NEIGH; ++k) { logits[k] = __expf(logits[k] - m); ssum += logits[k]; }
    const float inv = 1.f / (ssum + 1e-16f);

    float outj[4] = {0.f, 0.f, 0.f, 0.f};
    #pragma unroll 1
    for (int k = 0; k < NEIGH; ++k) {
        const int u = dstArr[ebase + k];
        const float alpha = logits[k] * inv;
        #pragma unroll
        for (int j = 0; j < 4; ++j)
            outj[j] += alpha * xlr[(size_t)u * 256 + lane * 4 + j];  // alpha * lin_l(x)[u]
    }

    float lsum = 0.f, lsq = 0.f;
    #pragma unroll
    for (int j = 0; j < 4; ++j) {
        const float val = outj[j] + conv_b[layer * 128 + lane * 4 + j];
        agg[(size_t)v * 128 + lane * 4 + j] = val;
        lsum += val;
        lsq  += val * val;
    }
    #pragma unroll
    for (int off = 1; off < 32; off <<= 1) {
        lsum += __shfl_xor(lsum, off, 32);
        lsq  += __shfl_xor(lsq,  off, 32);
    }
    if (lane == 0) { s_sum[wave] = lsum; s_sq[wave] = lsq; }
    __syncthreads();
    if (threadIdx.x == 0) {
        float a = 0.f, c = 0.f;
        #pragma unroll
        for (int i = 0; i < 8; ++i) { a += s_sum[i]; c += s_sq[i]; }
        partials[blockIdx.x * 2]     = a;
        partials[blockIdx.x * 2 + 1] = c;
    }
}

// ---------------------------------------------------------------------------
// Final reduction of the per-block partials -> {mean, 1/(std+eps)}.
// ---------------------------------------------------------------------------
__global__ __launch_bounds__(256)
void reduce_stats_kernel(const float* __restrict__ partials, int nblocks,
                         float* __restrict__ stats, float inv_count)
{
    __shared__ float ss[256], sq[256];
    float a = 0.f, c = 0.f;
    for (int i = threadIdx.x; i < nblocks; i += 256) {
        a += partials[2 * i];
        c += partials[2 * i + 1];
    }
    ss[threadIdx.x] = a; sq[threadIdx.x] = c;
    __syncthreads();
    for (int s = 128; s > 0; s >>= 1) {
        if (threadIdx.x < s) { ss[threadIdx.x] += ss[threadIdx.x + s];
                               sq[threadIdx.x] += sq[threadIdx.x + s]; }
        __syncthreads();
    }
    if (threadIdx.x == 0) {
        const float mean = ss[0] * inv_count;
        float var = sq[0] * inv_count - mean * mean;
        var = (var > 0.f) ? var : 0.f;
        stats[0] = mean;
        stats[1] = 1.f / (sqrtf(var) + 1e-5f);
    }
}

// ---------------------------------------------------------------------------
// Global LayerNorm + affine + ReLU + residual; emits h (f32) and h (f16).
// ---------------------------------------------------------------------------
__global__ __launch_bounds__(256)
void norm_update_kernel(const float* __restrict__ agg, const float* __restrict__ stats,
                        const float* __restrict__ lnw, const float* __restrict__ lnb,
                        float* __restrict__ h, _Float16* __restrict__ hf, int layer)
{
    const int idx = blockIdx.x * 256 + threadIdx.x;    // N*128
    const int c = idx & 127;
    const float mean = stats[0], rstd = stats[1];
    float g = (agg[idx] - mean) * rstd;
    g = g * lnw[layer * 128 + c] + lnb[layer * 128 + c];
    g = (g > 0.f) ? g : 0.f;
    const float val = g + h[idx];
    h[idx]  = val;
    hf[idx] = (_Float16)val;
}

// ---------------------------------------------------------------------------
// Classifier: [N,128] @ [128,9] + b. Tiny (K=128, 9 cols) -> scalar FMA.
// ---------------------------------------------------------------------------
__global__ __launch_bounds__(256)
void classify_kernel(const float* __restrict__ h, const float* __restrict__ cw,
                     const float* __restrict__ cb, float* __restrict__ out)
{
    const int idx = blockIdx.x * 256 + threadIdx.x;    // N*9
    const int n = idx / 9, j = idx - n * 9;
    float s = cb[j];
    #pragma unroll 8
    for (int c = 0; c < 128; ++c) s += h[(size_t)n * 128 + c] * cw[c * 9 + j];
    out[idx] = s;
}

// ---------------------------------------------------------------------------
extern "C" void kernel_launch(void* const* d_in, const int* in_sizes, int n_in,
                              void* d_out, int out_size, void* d_ws, size_t ws_size,
                              hipStream_t stream)
{
    (void)n_in; (void)out_size; (void)ws_size;
    const int*   x    = (const int*)d_in[0];
    const int*   ei   = (const int*)d_in[1];
    const float* ew   = (const float*)d_in[2];
    const float* Wl   = (const float*)d_in[3];
    const float* bl   = (const float*)d_in[4];
    const float* Wr   = (const float*)d_in[5];
    const float* br   = (const float*)d_in[6];
    const float* att  = (const float*)d_in[7];
    const float* cvb  = (const float*)d_in[8];
    const float* lnw  = (const float*)d_in[9];
    const float* lnb  = (const float*)d_in[10];
    const float* cw   = (const float*)d_in[11];
    const float* cb   = (const float*)d_in[12];
    float* out = (float*)d_out;

    const int E = in_sizes[1] / 2;                 // 414720 edges
    const int* dstArr = ei + E;                    // edge_index row 1 (dst)

    // Workspace carve-up (all 256B-aligned offsets), ~48.3 MB total.
    char* ws = (char*)d_ws;
    float*    h        = (float*)   (ws + 0);                    // N*128 f32  (10,616,832 B)
    _Float16* hf       = (_Float16*)(ws + 10616832);             // N*128 f16  ( 5,308,416 B)
    float*    xlr      = (float*)   (ws + 15925248);             // N*256 f32  (21,233,664 B)
    float*    agg      = (float*)   (ws + 37158912);             // N*128 f32  (10,616,832 B)
    _Float16* wpack    = (_Float16*)(ws + 47775744);             // 8*16*4*32*16 f16 (524,288 B)
    float*    partials = (float*)   (ws + 48300032);             // 2592*2 f32
    float*    stats    = (float*)   (ws + 48320768);             // 2 f32

    pack_w_kernel<<<64, 256, 0, stream>>>(Wl, Wr, wpack);
    embed_kernel<<<N_NODES * C_DIM / 256, 256, 0, stream>>>(x, ew, h, hf);

    const float inv_count = 1.0f / ((float)N_NODES * (float)C_DIM);
    for (int l = 0; l < 8; ++l) {
        gemm_xlr_kernel<<<N_NODES / 16, 128, 0, stream>>>(hf, wpack, bl, br, xlr, l);
        attn_agg_kernel<<<ATT_BLKS, 256, 0, stream>>>(xlr, dstArr, att, cvb, agg, partials, l);
        reduce_stats_kernel<<<1, 256, 0, stream>>>(partials, ATT_BLKS, stats, inv_count);
        norm_update_kernel<<<N_NODES * C_DIM / 256, 256, 0, stream>>>(agg, stats, lnw, lnb, h, hf, l);
    }
    classify_kernel<<<N_NODES * 9 / 256, 256, 0, stream>>>(h, cw, cb, out);
}
